// Attention_68667937128768
// MI455X (gfx1250) — compile-verified, matrix operand order checked by reading, not run on hardware
//
#include <hip/hip_runtime.h>
#include <hip/hip_bf16.h>

typedef __attribute__((ext_vector_type(16))) __bf16 v16bf;
typedef __attribute__((ext_vector_type(8)))  float  v8f;
typedef __attribute__((ext_vector_type(4)))  int    vi4;

#if defined(__AMDGCN__) && __has_builtin(__builtin_amdgcn_global_load_async_to_lds_b128)
#define HAS_ASYNC_LDS 1
#else
#define HAS_ASYNC_LDS 0
#endif

union Frag16 { uint4 u[2]; v16bf v; };

__device__ __forceinline__ v16bf load_frag(const __bf16* p0, const __bf16* p1) {
    Frag16 f;
    f.u[0] = *(const uint4*)p0;
    f.u[1] = *(const uint4*)p1;
    return f.v;
}

__device__ __forceinline__ v8f wmma_bf16(v16bf a, v16bf b, v8f c) {
    // D = A(16x32 bf16) x B(32x16 bf16) + C(16x16 f32)
    return __builtin_amdgcn_wmma_f32_16x16x32_bf16(
        false, a, false, b, (short)0, c, false, false);
}

#if HAS_ASYNC_LDS
typedef __attribute__((address_space(1))) vi4 gas_vi4;   // global (AS1)
typedef __attribute__((address_space(3))) vi4 las_vi4;   // LDS (AS3)

__device__ __forceinline__ void async_g2l_b128(const __bf16* gptr, __bf16* lptr) {
    // Per-lane 16B global -> LDS DMA (ASYNCcnt). AS3 offset = low 32 bits of
    // the generic address.
    __builtin_amdgcn_global_load_async_to_lds_b128(
        (gas_vi4*)(unsigned long long)(uintptr_t)gptr,
        (las_vi4*)(unsigned)(uintptr_t)lptr,
        0, 0);
}
__device__ __forceinline__ void wait_async0() {
#if __has_builtin(__builtin_amdgcn_s_wait_asynccnt)
    __builtin_amdgcn_s_wait_asynccnt(0);
#else
    asm volatile("s_wait_asynccnt 0" ::: "memory");
#endif
}
#endif

// ---------------------------------------------------------------------------
// fp32 -> bf16 conversion
// ---------------------------------------------------------------------------
__global__ void cvt_bf16(const float* __restrict__ in, __bf16* __restrict__ out, int n) {
    for (int i = blockIdx.x * blockDim.x + threadIdx.x; i < n; i += gridDim.x * blockDim.x)
        out[i] = (__bf16)in[i];
}

// ---------------------------------------------------------------------------
// NT GEMM: C[m,n] = sum_k A[m,k] * W[n,k]     (A: MxK bf16, W: NxK bf16)
// Block: 256 threads (8 waves as 4x2), tile 128(M) x 64(N), K-step 32 via LDS.
// Tile fill: 16B chunks, 4 per 32-elem row. A: 512 chunks (2/thread),
// B: 256 chunks (1/thread). Uses async global->LDS DMA when available.
// Requires M%128==0, N%64==0, K%32==0 (true for all launches here).
// ---------------------------------------------------------------------------
template <bool OUT_BF16>
__global__ __launch_bounds__(256) void gemm_bf16_nt(
    const __bf16* __restrict__ A, const __bf16* __restrict__ W,
    void* __restrict__ Cout, int M, int N, int K) {
    __shared__ __align__(16) __bf16 lA[128 * 32];
    __shared__ __align__(16) __bf16 lB[64 * 32];

    const int t    = threadIdx.x;
    const int lane = t & 31;
    const int wave = t >> 5;
    const int lr   = lane & 15;
    const int hf   = lane >> 4;
    const int m0   = blockIdx.y * 128;
    const int n0   = blockIdx.x * 64;
    const int wm   = (wave >> 1) * 32;   // wave's M offset in block tile
    const int wn   = (wave & 1) * 32;    // wave's N offset in block tile

    v8f acc[2][2] = {};

    const int row0 = t >> 2;             // 0..63
    const int seg  = (t & 3) * 8;        // element offset of 16B chunk in row

    for (int kt = 0; kt < K; kt += 32) {
        const __bf16* gA0 = A + (size_t)(m0 + row0) * K + kt + seg;
        const __bf16* gA1 = A + (size_t)(m0 + 64 + row0) * K + kt + seg;
        const __bf16* gB  = W + (size_t)(n0 + row0) * K + kt + seg;
        __bf16* sA0 = lA + row0 * 32 + seg;
        __bf16* sA1 = lA + (64 + row0) * 32 + seg;
        __bf16* sB  = lB + row0 * 32 + seg;
#if HAS_ASYNC_LDS
        async_g2l_b128(gA0, sA0);
        async_g2l_b128(gA1, sA1);
        async_g2l_b128(gB,  sB);
        wait_async0();
#else
        *(uint4*)sA0 = *(const uint4*)gA0;
        *(uint4*)sA1 = *(const uint4*)gA1;
        *(uint4*)sB  = *(const uint4*)gB;
#endif
        __syncthreads();

        v16bf af[2], bf2[2];
#pragma unroll
        for (int f = 0; f < 2; ++f) {
            const __bf16* pa = lA + (wm + f * 16 + lr) * 32;
            af[f] = load_frag(pa + hf * 8, pa + 16 + hf * 8);
            const __bf16* pb = lB + (wn + f * 16 + lr) * 32;
            bf2[f] = load_frag(pb + hf * 8, pb + 16 + hf * 8);
        }
#pragma unroll
        for (int fm = 0; fm < 2; ++fm)
#pragma unroll
            for (int fn = 0; fn < 2; ++fn)
                acc[fm][fn] = wmma_bf16(af[fm], bf2[fn], acc[fm][fn]);
        __syncthreads();
    }

#pragma unroll
    for (int fm = 0; fm < 2; ++fm) {
#pragma unroll
        for (int fn = 0; fn < 2; ++fn) {
            const int col = n0 + wn + fn * 16 + lr;
#pragma unroll
            for (int i = 0; i < 8; ++i) {
                const int row = m0 + wm + fm * 16 + hf * 8 + i;
                const float v = acc[fm][fn][i];
                if (OUT_BF16)
                    ((__bf16*)Cout)[(size_t)row * N + col] = (__bf16)v;
                else
                    ((float*)Cout)[(size_t)row * N + col] = v;
            }
        }
    }
}

// ---------------------------------------------------------------------------
// In-place RoPE on bf16 tensor laid out [B*S, heads, 64]; thread owns (d, d+32)
// ---------------------------------------------------------------------------
__global__ void rope_kernel(__bf16* __restrict__ t, const float* __restrict__ cosp,
                            const float* __restrict__ sinp, int heads, int total) {
    const int i = blockIdx.x * blockDim.x + threadIdx.x;
    if (i >= total) return;
    const int d   = i & 31;
    const int hh  = (i >> 5) % heads;
    const int row = i / (32 * heads);   // b*S + s
    const int s   = row & 2047;
    __bf16* base = t + ((size_t)row * heads + hh) * 64;
    const float c1  = cosp[s * 64 + d];
    const float sn1 = sinp[s * 64 + d];
    const float c2  = cosp[s * 64 + d + 32];
    const float sn2 = sinp[s * 64 + d + 32];
    const float x1 = (float)base[d];
    const float x2 = (float)base[d + 32];
    base[d]      = (__bf16)(x1 * c1 - x2 * sn1);
    base[d + 32] = (__bf16)(x2 * c2 + x1 * sn2);
}

// ---------------------------------------------------------------------------
// V transpose: [b, s, kvh, 64] -> [b, kvh, 64, s]   (bf16)
// ---------------------------------------------------------------------------
__global__ void vtrans_kernel(const __bf16* __restrict__ v, __bf16* __restrict__ vt) {
    const int i = blockIdx.x * blockDim.x + threadIdx.x;
    if (i >= (1 << 21)) return;            // 2*8*64*2048
    const int s  = i & 2047;
    const int d  = (i >> 11) & 63;
    const int kv = (i >> 17) & 7;
    const int b  = i >> 20;
    vt[i] = v[((size_t)((b * 2048 + s) * 8 + kv)) * 64 + d];
}

// ---------------------------------------------------------------------------
// Transposed flash attention. One wave per 16-query tile.
//   S^T = K_tile(16x64) @ Q^T(64x16)  -> C frag: lane holds query (col),
//   keys live in-register  => per-query softmax stats are one scalar/lane.
//   P^T C-frag repacks directly into the B operand of O^T = V^T @ P^T.
// Q/K laid out [b, s, h, 64] bf16; VT is [b, kvh, 64, S] bf16.
// O written [b, s, h, 64] bf16.
// ---------------------------------------------------------------------------
__global__ __launch_bounds__(128) void attn_kernel(
    const __bf16* __restrict__ Q, const __bf16* __restrict__ Kc,
    const __bf16* __restrict__ VT, __bf16* __restrict__ O) {
    const int S    = 2048;
    const int lane = threadIdx.x & 31;
    const int wave = threadIdx.x >> 5;
    const int lr   = lane & 15;
    const int hf   = lane >> 4;
    const int qt   = blockIdx.x * 4 + wave;
    const int h    = blockIdx.y;
    const int b    = blockIdx.z;
    const int kvh  = h >> 2;               // n_rep = 4
    const int q0   = qt * 16;
    const int qidx = q0 + lr;

    // Q fragments as B operand of S^T: lane holds query lr's dims
    const __bf16* qrow = Q + ((size_t)(b * S + qidx) * 32 + h) * 64;
    const v16bf qb_lo = load_frag(qrow + hf * 8,      qrow + 16 + hf * 8);
    const v16bf qb_hi = load_frag(qrow + 32 + hf * 8, qrow + 48 + hf * 8);

    v8f acc[4] = {};                       // O^T: 4 x (16 dims x 16 queries)
    float m_run = -__builtin_inff();
    float l_run = 0.f;

    const __bf16* vbase = VT + ((size_t)(b * 8 + kvh) * 64) * S;

    for (int kb = 0; kb < q0 + 16; kb += 32) {
        v8f s[2];
#pragma unroll
        for (int g = 0; g < 2; ++g) {      // key groups kb.. / kb+16..
            const __bf16* krow =
                Kc + ((size_t)(b * S + kb + g * 16 + lr) * 8 + kvh) * 64;
            v16bf ka_lo = load_frag(krow + hf * 8,      krow + 16 + hf * 8);
            v16bf ka_hi = load_frag(krow + 32 + hf * 8, krow + 48 + hf * 8);
            v8f z = {};
            z    = wmma_bf16(ka_lo, qb_lo, z);
            s[g] = wmma_bf16(ka_hi, qb_hi, z);
        }

        // scale + causal mask + block max (keys in-register; one xor-16 reduce)
        float mb = -__builtin_inff();
#pragma unroll
        for (int g = 0; g < 2; ++g)
#pragma unroll
            for (int i = 0; i < 8; ++i) {
                const int key = kb + g * 16 + hf * 8 + i;
                float v = s[g][i] * 0.125f;          // 1/sqrt(64)
                v = (key <= qidx) ? v : -__builtin_inff();
                s[g][i] = v;
                mb = fmaxf(mb, v);
            }
        mb = fmaxf(mb, __shfl_xor(mb, 16, 32));
        const float m_new = fmaxf(m_run, mb);
        const float corr  = __expf(m_run - m_new);   // 0 on first block
        m_run = m_new;
        l_run *= corr;

        // exp + pack P^T straight into the B fragment (no cross-lane moves)
        v16bf pb;
#pragma unroll
        for (int g = 0; g < 2; ++g)
#pragma unroll
            for (int i = 0; i < 8; ++i) {
                const float p = __expf(s[g][i] - m_new);
                l_run += p;
                pb[g * 8 + i] = (__bf16)p;
            }

        // O^T += V^T_tile(16x32) @ P^T(32x16)
#pragma unroll
        for (int df = 0; df < 4; ++df) {
#pragma unroll
            for (int i = 0; i < 8; ++i) acc[df][i] *= corr;
            const __bf16* vrow = vbase + (size_t)(df * 16 + lr) * S + kb;
            v16bf va = load_frag(vrow + hf * 8, vrow + 16 + hf * 8);
            acc[df] = wmma_bf16(va, pb, acc[df]);
        }
    }

    const float l_tot = l_run + __shfl_xor(l_run, 16, 32);
    const float inv   = 1.f / l_tot;
    __bf16* orow = O + ((size_t)(b * S + qidx) * 32 + h) * 64;
#pragma unroll
    for (int df = 0; df < 4; ++df)
#pragma unroll
        for (int i = 0; i < 8; ++i)
            orow[df * 16 + hf * 8 + i] = (__bf16)(acc[df][i] * inv);
}

// ---------------------------------------------------------------------------
// Host-side orchestration
// ---------------------------------------------------------------------------
extern "C" void kernel_launch(void* const* d_in, const int* in_sizes, int n_in,
                              void* d_out, int out_size, void* d_ws, size_t ws_size,
                              hipStream_t stream) {
    const float* x    = (const float*)d_in[0];   // [2, 2048, 2048]
    const float* cosp = (const float*)d_in[1];   // [2048, 64]
    const float* sinp = (const float*)d_in[2];   // [2048, 64]
    const float* wq   = (const float*)d_in[3];   // [2048, 2048]
    const float* wk   = (const float*)d_in[4];   // [512, 2048]
    const float* wv   = (const float*)d_in[5];   // [512, 2048]
    const float* wo   = (const float*)d_in[6];   // [2048, 2048]
    float* out = (float*)d_out;                  // [2, 2048, 2048]

    char* ws = (char*)d_ws;
    __bf16* xb    = (__bf16*)(ws);               // 16 MB  x bf16     [4096,2048]
    __bf16* wqb   = (__bf16*)(ws + 16777216);    //  8 MB
    __bf16* wkb   = (__bf16*)(ws + 25165824);    //  2 MB
    __bf16* wvb   = (__bf16*)(ws + 27262976);    //  2 MB
    __bf16* wob   = (__bf16*)(ws + 29360128);    //  8 MB
    __bf16* qb    = (__bf16*)(ws + 37748736);    // 16 MB  q (roped)  [4096,32,64]
    __bf16* kbuf  = (__bf16*)(ws + 54525952);    //  4 MB  k (roped)  [4096,8,64]
    __bf16* vbuf  = (__bf16*)(ws + 58720256);    //  4 MB  v          [4096,8,64]
    __bf16* vtb   = (__bf16*)(ws + 62914560);    //  4 MB  v^T        [2,8,64,2048]
    __bf16* attnb = (__bf16*)(ws + 67108864);    // 16 MB  attn out   [4096,2048]

    // fp32 -> bf16
    cvt_bf16<<<2048, 256, 0, stream>>>(x,  xb,  8388608);
    cvt_bf16<<<1024, 256, 0, stream>>>(wq, wqb, 4194304);
    cvt_bf16<<<256,  256, 0, stream>>>(wk, wkb, 1048576);
    cvt_bf16<<<256,  256, 0, stream>>>(wv, wvb, 1048576);
    cvt_bf16<<<1024, 256, 0, stream>>>(wo, wob, 4194304);

    // Q/K/V projections (y = x @ W^T), bf16 outputs
    gemm_bf16_nt<true><<<dim3(32, 32), 256, 0, stream>>>(xb, wqb, qb,   4096, 2048, 2048);
    gemm_bf16_nt<true><<<dim3(8,  32), 256, 0, stream>>>(xb, wkb, kbuf, 4096, 512,  2048);
    gemm_bf16_nt<true><<<dim3(8,  32), 256, 0, stream>>>(xb, wvb, vbuf, 4096, 512,  2048);

    // RoPE (in-place) + V transpose
    rope_kernel<<<16384, 256, 0, stream>>>(qb,   cosp, sinp, 32, 4096 * 32 * 32);
    rope_kernel<<<4096,  256, 0, stream>>>(kbuf, cosp, sinp, 8,  4096 * 8 * 32);
    vtrans_kernel<<<8192, 256, 0, stream>>>(vbuf, vtb);

    // Causal flash attention (transposed formulation)
    attn_kernel<<<dim3(32, 32, 2), 128, 0, stream>>>(qb, kbuf, vtb, attnb);

    // Output projection -> fp32 d_out
    gemm_bf16_nt<false><<<dim3(32, 32), 256, 0, stream>>>(attnb, wob, out, 4096, 2048, 2048);
}
